// DynamicConvultionAttention_67250597920857
// MI455X (gfx1250) — matrix operand order repaired and checked by reading, not compiled, not run on hardware
//
#include <hip/hip_runtime.h>
#include <hip/hip_fp16.h>
#include <math.h>

typedef __attribute__((ext_vector_type(16))) _Float16 v16h;
typedef __attribute__((ext_vector_type(8)))  _Float16 v8h;
typedef __attribute__((ext_vector_type(8)))  float    v8f;

#define B_  64
#define T_  8192
#define DC_ 8
#define DK_ 21
#define SC_ 8
#define SK_ 21
#define PL_ 11
#define H_  128
#define R_  256
#define TT  128          // t-positions per block
#define HALO ((DK_ - 1) / 2)

// Branch-free fast tanh: native v_tanh_f32 on gfx1250 if the builtin exists,
// else exp/rcp formulation (v_exp_f32 + v_rcp_f32, no control flow).
__device__ __forceinline__ float fast_tanh(float x) {
#if __has_builtin(__builtin_amdgcn_tanhf)
  return __builtin_amdgcn_tanhf(x);
#elif __has_builtin(__builtin_amdgcn_tanh_f32)
  return __builtin_amdgcn_tanh_f32(x);
#else
  float t = __expf(2.0f * x);
  return 1.0f - 2.0f * __builtin_amdgcn_rcpf(t + 1.0f);
#endif
}

// ---------------------------------------------------------------------------
// Kernel 1: per-batch dynamic kernels  G[b, c, k] = (tanh(s W^T + b) V^T)
// Tiny (64 x 168 outputs) -> plain VALU kernel, one block per batch row.
// ---------------------------------------------------------------------------
__global__ void dca_dynamic_kernels(const float* __restrict__ s,
                                    const float* __restrict__ W_w,
                                    const float* __restrict__ W_b,
                                    const float* __restrict__ V_w,
                                    float* __restrict__ G) {
  __shared__ float sbuf[R_];
  __shared__ float hbuf[H_];
  const int b   = blockIdx.x;
  const int tid = threadIdx.x;   // blockDim.x == 128

  for (int i = tid; i < R_; i += blockDim.x) sbuf[i] = s[b * R_ + i];
  __syncthreads();

  float acc = W_b[tid];
  #pragma unroll 4
  for (int r = 0; r < R_; ++r) acc += sbuf[r] * W_w[tid * R_ + r];
  hbuf[tid] = fast_tanh(acc);
  __syncthreads();

  for (int j = tid; j < DC_ * DK_; j += blockDim.x) {
    float g = 0.f;
    #pragma unroll 4
    for (int h = 0; h < H_; ++h) g += hbuf[h] * V_w[j * H_ + h];
    G[b * (DC_ * DK_) + j] = g;
  }
}

// ---------------------------------------------------------------------------
// Kernel 2: fused scores.  One block = one batch row x 128 t-positions.
//   - alpha tile (+/- 10 halo) -> LDS
//   - 8 static + 8 dynamic 21-tap convs -> f16 feature tile X[128][16]
//   - prior: causal 11-tap conv + log  -> pLDS
//   - 8x WMMA f32_16x16x32_f16 per wave (fully unrolled over H),
//     K padded 16->32 via zeroed LDS rows -> uniform, branch-free fragments
//   - C preloaded with bias; D -> v_tanh -> * v_w[h] -> shfl_xor reduce
// ---------------------------------------------------------------------------
__global__ void dca_scores(const float* __restrict__ alpha,
                           const float* __restrict__ F_w,
                           const float* __restrict__ U_w,
                           const float* __restrict__ T_w,
                           const float* __restrict__ T_b,
                           const float* __restrict__ v_w,
                           const float* __restrict__ P_flip,
                           const float* __restrict__ G,
                           float* __restrict__ e_out) {
  __shared__ float    alphaS[TT + DK_ - 1];          // alpha tile + halo
  __shared__ float    Kc[16][DK_];                   // combined conv kernels
  __shared__ float    Pf[PL_];                       // prior taps
  __shared__ float    pLDS[TT];                      // log-prior per t
  __shared__ float    tbLDS[H_];                     // bias
  __shared__ float    vwLDS[H_];                     // projection vector
  __shared__ __align__(16) _Float16 X[TT][16];       // feature tile (f16)
  __shared__ __align__(16) _Float16 WcT[H_][32];     // weights, K-padded to 32

  const int tilesPerRow = T_ / TT;
  const int b   = blockIdx.x / tilesPerRow;
  const int t0  = (blockIdx.x % tilesPerRow) * TT;
  const int tid = threadIdx.x;                       // blockDim.x == 256

  // ---- stage alpha tile with halo (zero-padded at row ends) ----
  for (int i = tid; i < TT + DK_ - 1; i += blockDim.x) {
    int tg = t0 + i - HALO;
    alphaS[i] = (tg >= 0 && tg < T_) ? alpha[b * T_ + tg] : 0.f;
  }
  // ---- stage conv kernels: channels 0..7 static, 8..15 dynamic ----
  for (int i = tid; i < 16 * DK_; i += blockDim.x) {
    int c = i / DK_, k = i % DK_;
    Kc[c][k] = (c < SC_) ? F_w[c * SK_ + k]
                         : G[b * (DC_ * DK_) + (c - SC_) * DK_ + k];
  }
  if (tid < PL_) Pf[tid] = P_flip[tid];
  if (tid < H_) { tbLDS[tid] = T_b[tid]; vwLDS[tid] = v_w[tid]; }
  // ---- stage transposed weight matrix, K-contiguous, rows 16..31 = 0 ----
  // WcT[h][c] = c<8 ? U_w[h][c] : (c<16 ? T_w[h][c-8] : 0)
  for (int i = tid; i < H_ * 32; i += blockDim.x) {
    int h = i >> 5, c = i & 31;
    float w = (c < SC_) ? U_w[h * SC_ + c]
            : (c < 16) ? T_w[h * DC_ + (c - SC_)]
                       : 0.f;
    WcT[h][c] = (_Float16)w;
  }
  __syncthreads();

  // ---- features: X[t][c] = SAME-conv(alpha, Kc[c]) ----
  for (int idx = tid; idx < TT * 16; idx += blockDim.x) {
    int t = idx >> 4, c = idx & 15;
    float acc = 0.f;
    #pragma unroll
    for (int k = 0; k < DK_; ++k) acc += alphaS[t + k] * Kc[c][k];
    X[t][c] = (_Float16)acc;
  }
  // ---- prior: p[t] = log(clip(causal-conv(alpha, P_flip))) ----
  for (int t = tid; t < TT; t += blockDim.x) {
    float acc = 0.f;
    #pragma unroll
    for (int k = 0; k < PL_; ++k) acc += alphaS[t + k] * Pf[k];
    pLDS[t] = logf(fmaxf(acc, 1e-6f));
  }
  __syncthreads();

  // ---- WMMA stage: each wave owns rows [wave*16, wave*16+16) ----
  const int wave = tid >> 5;       // 0..7
  const int lane = tid & 31;
  const int half = lane >> 4;      // 0 or 1
  const int l16  = lane & 15;

  // A fragment (16x32 f16): both lane-halves hold row M = l16;
  // VGPR0-3 carry K = i + 8*half (one 16B ds_load_b128), K>=16 zero pad.
  v16h afrag;
  {
    const int m = wave * 16 + l16;
    const v8h alo = *(const v8h*)&X[m][8 * half];
    #pragma unroll
    for (int i = 0; i < 8; ++i) afrag[i] = alo[i];
    #pragma unroll
    for (int i = 8; i < 16; ++i) afrag[i] = (_Float16)0;
  }

  float acc[8];
  #pragma unroll
  for (int r = 0; r < 8; ++r) acc[r] = 0.f;

  #pragma unroll
  for (int nt = 0; nt < H_ / 16; ++nt) {
    const int n0 = nt * 16;

    // B fragment (32x16 f16): element i<8 is K = i + 8*half, i>=8 is
    // K = 16 + 8*half + (i-8), for column N = n0 + l16.  Rows 16..31 of WcT
    // are zero pad; two 16B ds_load_b128 per lane.
    const v8h* bp = (const v8h*)&WcT[n0 + l16][0];
    const v8h b0 = bp[half];
    const v8h b1 = bp[2 + half];
    v16h bfrag;
    #pragma unroll
    for (int i = 0; i < 8; ++i) { bfrag[i] = b0[i]; bfrag[8 + i] = b1[i]; }

    // C init = bias T_b[n] (constant down each column of the 16x16 tile)
    const float tb = tbLDS[n0 + l16];
    v8f c;
    #pragma unroll
    for (int r = 0; r < 8; ++r) c[r] = tb;

    c = __builtin_amdgcn_wmma_f32_16x16x32_f16(
        /*neg_a=*/false, afrag, /*neg_b=*/false, bfrag,
        /*c_mod=*/(short)0, c, /*reuse_a=*/false, /*reuse_b=*/false);

    // D layout: element r, lane L -> row M = r + 8*(L>=16), col N = n0 + L%16
    const float vv = vwLDS[n0 + l16];
    #pragma unroll
    for (int r = 0; r < 8; ++r) acc[r] += fast_tanh(c[r]) * vv;
  }

  // Reduce over the 16 column-lanes within each half-wave (masks < 16).
  #pragma unroll
  for (int r = 0; r < 8; ++r) {
    float v = acc[r];
    #pragma unroll
    for (int off = 1; off < 16; off <<= 1) v += __shfl_xor(v, off, 32);
    acc[r] = v;
  }
  if (l16 == 0) {
    #pragma unroll
    for (int r = 0; r < 8; ++r) {
      const int tl = wave * 16 + half * 8 + r;   // matches C/D M layout
      e_out[b * T_ + t0 + tl] = acc[r] + pLDS[tl];
    }
  }
}

// ---------------------------------------------------------------------------
// Kernel 3: row softmax over T=8192 (one block per batch row)
// ---------------------------------------------------------------------------
__global__ void dca_softmax(const float* __restrict__ e,
                            float* __restrict__ out) {
  __shared__ float red[256];
  const int b   = blockIdx.x;
  const int tid = threadIdx.x;  // 256
  const float* row = e + b * T_;

  float m = -INFINITY;
  for (int t = tid; t < T_; t += 256) m = fmaxf(m, row[t]);
  red[tid] = m;
  __syncthreads();
  for (int s = 128; s > 0; s >>= 1) {
    if (tid < s) red[tid] = fmaxf(red[tid], red[tid + s]);
    __syncthreads();
  }
  m = red[0];
  __syncthreads();

  float sum = 0.f;
  for (int t = tid; t < T_; t += 256) sum += __expf(row[t] - m);
  red[tid] = sum;
  __syncthreads();
  for (int s = 128; s > 0; s >>= 1) {
    if (tid < s) red[tid] += red[tid + s];
    __syncthreads();
  }
  const float inv = 1.f / red[0];
  for (int t = tid; t < T_; t += 256)
    out[b * T_ + t] = __expf(row[t] - m) * inv;
}

// ---------------------------------------------------------------------------
extern "C" void kernel_launch(void* const* d_in, const int* in_sizes, int n_in,
                              void* d_out, int out_size, void* d_ws, size_t ws_size,
                              hipStream_t stream) {
  const float* s      = (const float*)d_in[0];
  const float* alpha  = (const float*)d_in[1];
  const float* W_w    = (const float*)d_in[2];
  const float* W_b    = (const float*)d_in[3];
  const float* V_w    = (const float*)d_in[4];
  const float* F_w    = (const float*)d_in[5];
  const float* U_w    = (const float*)d_in[6];
  const float* T_w    = (const float*)d_in[7];
  const float* T_b    = (const float*)d_in[8];
  const float* v_w    = (const float*)d_in[9];
  const float* P_flip = (const float*)d_in[10];

  float* e_ws = (float*)d_ws;               // B*T scores
  float* G_ws = e_ws + (size_t)B_ * T_;     // B*DC*DK dynamic kernels

  dca_dynamic_kernels<<<B_, 128, 0, stream>>>(s, W_w, W_b, V_w, G_ws);
  dca_scores<<<B_ * (T_ / TT), 256, 0, stream>>>(alpha, F_w, U_w, T_w, T_b,
                                                 v_w, P_flip, G_ws, e_ws);
  dca_softmax<<<B_, 256, 0, stream>>>(e_ws, (float*)d_out);
}
